// IdeficsPerceiverResampler_55327768707752
// MI455X (gfx1250) — compile-verified, matrix operand order checked by reading, not run on hardware
//
#include <hip/hip_runtime.h>
#include <hip/hip_bf16.h>
#include <math.h>

// ---------------------------------------------------------------------------
// Types for WMMA fragments (wave32, v_wmma_f32_16x16x32_f16)
// ---------------------------------------------------------------------------
typedef _Float16 half8  __attribute__((ext_vector_type(8)));
typedef _Float16 half16 __attribute__((ext_vector_type(16)));
typedef float    f32x8  __attribute__((ext_vector_type(8)));

// Load a 16-half fragment as two contiguous 16-byte chunks (chunk1 at +16 halves).
// Matches the CDNA5 16-bit A/B layout: lane<16 holds K{0..7,16..23},
// lane>=16 holds K{8..15,24..31}; caller adds the hi*8 offset.
__device__ __forceinline__ half16 ld_frag(const _Float16* p) {
  half8 a = *(const half8*)(p);
  half8 b = *(const half8*)(p + 16);
  return __builtin_shufflevector(a, b, 0,1,2,3,4,5,6,7,8,9,10,11,12,13,14,15);
}

__device__ __forceinline__ f32x8 wmma_f16(half16 a, half16 b, f32x8 c) {
  return __builtin_amdgcn_wmma_f32_16x16x32_f16(false, a, false, b, (short)0, c,
                                                false, false);
}

// Async global -> LDS (DMA, tracked by ASYNCcnt; no VGPR round-trip).
// VDST = LDS byte address (low 32 bits of the flat shared pointer),
// VADDR = 64-bit global address.  CDNA5 ISA.
__device__ __forceinline__ void async_g2l_b128(const void* gptr, void* lptr) {
  unsigned int loff = (unsigned int)(unsigned long long)lptr;
  asm volatile("global_load_async_to_lds_b128 %0, %1, off"
               :
               : "v"(loff), "v"(gptr)
               : "memory");
}

__device__ __forceinline__ void wait_async_le4() {
  asm volatile("s_wait_asynccnt 0x4" ::: "memory");
}
__device__ __forceinline__ void wait_async_0() {
  asm volatile("s_wait_asynccnt 0x0" ::: "memory");
}

// ---------------------------------------------------------------------------
// Problem constants
// ---------------------------------------------------------------------------
#define BB 32
#define SS 2048
#define DD 1280
#define HH 16
#define HD 96
#define NLAT 64
#define KVROWS 2112            // S + NL
#define HHD 1536               // H * HD
#define FFD 5120

// ---------------------------------------------------------------------------
// lat init: broadcast latents [64,1280] -> lat [32*64,1280] f32
// ---------------------------------------------------------------------------
__global__ void lat_init_kernel(float* __restrict__ lat,
                                const float* __restrict__ latents) {
  size_t idx = (size_t)blockIdx.x * 256 + threadIdx.x;
  if (idx >= (size_t)BB * NLAT * DD) return;
  size_t d = idx % DD;
  size_t n = (idx / DD) % NLAT;
  lat[idx] = latents[n * DD + d];
}

// ---------------------------------------------------------------------------
// Weight transpose+convert: W f32 [K][N] -> WT f16 [N][K]
// ---------------------------------------------------------------------------
__global__ void wtrans_kernel(const float* __restrict__ W,
                              _Float16* __restrict__ WT, int K, int N) {
  size_t idx = (size_t)blockIdx.x * 256 + threadIdx.x;
  if (idx >= (size_t)K * N) return;
  int n = (int)(idx % N);
  int k = (int)(idx / N);
  WT[(size_t)n * K + k] = (_Float16)W[idx];
}

// ---------------------------------------------------------------------------
// Row LayerNorm over D=1280.  One row per 256-thread block.
// dst row = (r / perBsrc) * perBdst + dstOff + (r % perBsrc)
// Writes f16 (dsth) or f32 (dstf).
// ---------------------------------------------------------------------------
__global__ __launch_bounds__(256) void ln1280_kernel(
    const float* __restrict__ src, const float* __restrict__ g,
    const float* __restrict__ bta, _Float16* __restrict__ dsth,
    float* __restrict__ dstf, int perBsrc, int perBdst, int dstOff) {
  __shared__ float red[2][8];
  int r = blockIdx.x;
  int tid = threadIdx.x;
  const float* row = src + (size_t)r * DD;
  float x[5];
  float s = 0.f, q = 0.f;
#pragma unroll
  for (int i = 0; i < 5; ++i) {
    x[i] = row[tid + 256 * i];
    s += x[i];
    q += x[i] * x[i];
  }
#pragma unroll
  for (int m = 16; m; m >>= 1) {
    s += __shfl_xor(s, m, 32);
    q += __shfl_xor(q, m, 32);
  }
  int wave = tid >> 5, lane = tid & 31;
  if (lane == 0) { red[0][wave] = s; red[1][wave] = q; }
  __syncthreads();
  s = 0.f; q = 0.f;
#pragma unroll
  for (int w = 0; w < 8; ++w) { s += red[0][w]; q += red[1][w]; }
  float mean = s * (1.f / DD);
  float var  = q * (1.f / DD) - mean * mean;
  float inv  = rsqrtf(var + 1e-5f);
  int bi = r / perBsrc, ri = r % perBsrc;
  size_t drow = (size_t)bi * perBdst + dstOff + ri;
#pragma unroll
  for (int i = 0; i < 5; ++i) {
    int c = tid + 256 * i;
    float y = (x[i] - mean) * inv * g[c] + bta[c];
    if (dsth) dsth[drow * DD + c] = (_Float16)y;
    else      dstf[drow * DD + c] = y;
  }
}

// ---------------------------------------------------------------------------
// Per-head q LayerNorm over HD=96 + qk_scale, f32 -> f16 (same layout).
// One row (token,head) per wave32.
// ---------------------------------------------------------------------------
__global__ __launch_bounds__(256) void qln_kernel(
    const float* __restrict__ q, const float* __restrict__ g,
    const float* __restrict__ bta, _Float16* __restrict__ out, float scale,
    int nrows) {
  int rid = blockIdx.x * 8 + (threadIdx.x >> 5);
  if (rid >= nrows) return;
  int lane = threadIdx.x & 31;
  int tok = rid >> 4, h = rid & 15;
  const float* p = q + (size_t)tok * HHD + h * HD;
  float x[3];
  float s = 0.f, ss = 0.f;
#pragma unroll
  for (int i = 0; i < 3; ++i) {
    x[i] = p[lane * 3 + i];
    s += x[i];
    ss += x[i] * x[i];
  }
#pragma unroll
  for (int m = 16; m; m >>= 1) {
    s += __shfl_xor(s, m, 32);
    ss += __shfl_xor(ss, m, 32);
  }
  float mean = s * (1.f / HD);
  float inv = rsqrtf(ss * (1.f / HD) - mean * mean + 1e-5f);
  _Float16* o = out + (size_t)tok * HHD + h * HD;
#pragma unroll
  for (int i = 0; i < 3; ++i) {
    int c = lane * 3 + i;
    o[c] = (_Float16)(((x[i] - mean) * inv * g[c] + bta[c]) * scale);
  }
}

// ---------------------------------------------------------------------------
// Tiled f16 WMMA GEMM: C[M][N] = A[M][K] * BT[N][K]^T
// Block tile 128x128x32, 8 waves (4m x 2n), each wave 32x64 via 2x4 WMMA accs.
// Double-buffered LDS fed by async global->LDS DMA (ASYNCcnt pipelined):
// tile t+1 is in flight while tile t's WMMAs execute.
// mode: 0 = f32 store, 1 = f16 store, 2 = relu -> f16, 3 = f32 store + Res add
// All of M%128==0, N%128==0, K%32==0 hold for every call site.
// ---------------------------------------------------------------------------
#define GBM 128
#define GBN 128
#define GBK 32

__global__ __launch_bounds__(256) void gemm_f16_wmma_kernel(
    const _Float16* __restrict__ A, const _Float16* __restrict__ BT,
    float* __restrict__ Cf, _Float16* __restrict__ Ch,
    const float* __restrict__ Res, int M, int N, int K, int mode) {
  __shared__ __align__(16) _Float16 As[2][GBM][GBK];
  __shared__ __align__(16) _Float16 Bs[2][GBN][GBK];
  int tid = threadIdx.x;
  int lane = tid & 31, wave = tid >> 5;
  int hi = (lane >> 4) & 1, l15 = lane & 15;
  int wm = wave >> 1, wn = wave & 1;
  int blockM = blockIdx.y * GBM, blockN = blockIdx.x * GBN;

  // this thread's two 16B chunks within the 128x32 tile
  int r0   = (tid * 2) >> 2;            // chunk 0 row
  int c0   = ((tid * 2) & 3) * 8;       // chunk 0 col
  int r1   = (tid * 2 + 1) >> 2;
  int c1   = ((tid * 2 + 1) & 3) * 8;

  // issue async DMA for one K-tile into buffer `buf` (4 ops/thread)
  auto issue_tile = [&](int buf, int kt) {
    async_g2l_b128(&A [(size_t)(blockM + r0) * K + kt + c0], &As[buf][r0][c0]);
    async_g2l_b128(&BT[(size_t)(blockN + r0) * K + kt + c0], &Bs[buf][r0][c0]);
    async_g2l_b128(&A [(size_t)(blockM + r1) * K + kt + c1], &As[buf][r1][c1]);
    async_g2l_b128(&BT[(size_t)(blockN + r1) * K + kt + c1], &Bs[buf][r1][c1]);
  };

  f32x8 zero = {};
  f32x8 acc[2][4];
#pragma unroll
  for (int i = 0; i < 2; ++i)
#pragma unroll
    for (int j = 0; j < 4; ++j) acc[i][j] = zero;

  int nk = K / GBK;
  issue_tile(0, 0);

  for (int t = 0; t < nk; ++t) {
    int buf = t & 1;
    if (t + 1 < nk) {
      issue_tile(buf ^ 1, (t + 1) * GBK);
      wait_async_le4();   // async completes in order: tile t's 4 ops are done
    } else {
      wait_async_0();
    }
    __syncthreads();      // all waves' DMA for tile t has landed in LDS

    half16 af[2], bf[4];
#pragma unroll
    for (int i = 0; i < 2; ++i)
      af[i] = ld_frag(&As[buf][wm * 32 + i * 16 + l15][hi * 8]);
#pragma unroll
    for (int j = 0; j < 4; ++j)
      bf[j] = ld_frag(&Bs[buf][wn * 64 + j * 16 + l15][hi * 8]);
#pragma unroll
    for (int i = 0; i < 2; ++i)
#pragma unroll
      for (int j = 0; j < 4; ++j)
        acc[i][j] = wmma_f16(af[i], bf[j], acc[i][j]);
    __syncthreads();      // done reading buf before it is DMA-overwritten
  }

  // epilogue: D fragment element (row = v + 8*hi, col = lane&15)
#pragma unroll
  for (int i = 0; i < 2; ++i)
#pragma unroll
    for (int j = 0; j < 4; ++j) {
      int row0 = blockM + wm * 32 + i * 16 + 8 * hi;
      int col = blockN + wn * 64 + j * 16 + l15;
#pragma unroll
      for (int v = 0; v < 8; ++v) {
        size_t idx = (size_t)(row0 + v) * N + col;
        float x = acc[i][j][v];
        if (mode == 0)      Cf[idx] = x;
        else if (mode == 1) Ch[idx] = (_Float16)x;
        else if (mode == 2) Ch[idx] = (_Float16)(x > 0.f ? x : 0.f);
        else                Cf[idx] = x + Res[idx];
      }
    }
}

// ---------------------------------------------------------------------------
// Flash-style attention, one (b,h) per 128-thread block (4 waves).
// Q f16 [B*64][1536] (already LN'ed + scaled), K f16 (k-LN fused here),
// V f16, output f16 [B*64][1536].
// Wave w owns query rows 16w..16w+15; 64-key tiles; WMMA for QK^T and P*V.
// ---------------------------------------------------------------------------
__global__ __launch_bounds__(128) void attn_kernel(
    const _Float16* __restrict__ Q, const _Float16* __restrict__ Kd,
    const _Float16* __restrict__ Vd, const float* __restrict__ kg,
    const float* __restrict__ kb, _Float16* __restrict__ O) {
  __shared__ __align__(16) _Float16 Kt[64][HD];      // keys x dims (k-LN'ed)
  __shared__ __align__(16) _Float16 VT[HD][64];      // dims x keys
  __shared__ __align__(16) _Float16 P[4][16][64];    // per-wave probabilities
  __shared__ float kgs[HD], kbs[HD];

  int b = blockIdx.x >> 4, h = blockIdx.x & 15;
  int tid = threadIdx.x, lane = tid & 31, wave = tid >> 5;
  int hi = (lane >> 4) & 1, l15 = lane & 15;
  if (tid < HD) { kgs[tid] = kg[tid]; kbs[tid] = kb[tid]; }

  // q fragments for this wave's 16 rows: K = 96 -> 3 chunks of 32
  half16 qf[3];
  {
    const _Float16* qrow =
        Q + (size_t)(b * NLAT + wave * 16 + l15) * HHD + h * HD;
#pragma unroll
    for (int kc = 0; kc < 3; ++kc) qf[kc] = ld_frag(qrow + kc * 32 + hi * 8);
  }

  f32x8 zero = {};
  f32x8 o[6];
#pragma unroll
  for (int n = 0; n < 6; ++n) o[n] = zero;
  float rm[8], rs[8];
#pragma unroll
  for (int v = 0; v < 8; ++v) { rm[v] = -1e30f; rs[v] = 0.f; }

  for (int jt = 0; jt < KVROWS; jt += 64) {
    __syncthreads();  // LDS reuse fence (also covers kgs/kbs init on iter 0)

    // ---- stage K tile with fused per-head LayerNorm (one key per thread) ----
    if (tid < 64) {
      const _Float16* krow =
          Kd + (size_t)(b * KVROWS + jt + tid) * HHD + h * HD;
      float s = 0.f, q2 = 0.f;
      for (int d = 0; d < HD; ++d) {
        float xx = (float)krow[d];
        s += xx; q2 += xx * xx;
      }
      float mean = s * (1.f / HD);
      float inv = rsqrtf(q2 * (1.f / HD) - mean * mean + 1e-5f);
      for (int d = 0; d < HD; ++d) {
        float xx = (float)krow[d];
        Kt[tid][d] = (_Float16)((xx - mean) * inv * kgs[d] + kbs[d]);
      }
    }
    // ---- stage V tile transposed: VT[dim][key] ----
    {
      int key = tid >> 1;
      int d0 = (tid & 1) * 48;
      const _Float16* vrow =
          Vd + (size_t)(b * KVROWS + jt + key) * HHD + h * HD + d0;
      for (int d = 0; d < 48; ++d) VT[d0 + d][key] = vrow[d];
    }
    __syncthreads();

    // ---- scores: 16 x 64 per wave via WMMA ----
    f32x8 sc[4];
#pragma unroll
    for (int nt = 0; nt < 4; ++nt) {
      f32x8 c = zero;
#pragma unroll
      for (int kc = 0; kc < 3; ++kc) {
        half16 kf = ld_frag(&Kt[nt * 16 + l15][kc * 32 + hi * 8]);
        c = wmma_f16(qf[kc], kf, c);
      }
      sc[nt] = c;
    }

    // ---- streaming softmax: row = v + 8*hi, 16-lane row reductions ----
#pragma unroll
    for (int v = 0; v < 8; ++v) {
      float m = sc[0][v];
      m = fmaxf(m, sc[1][v]);
      m = fmaxf(m, sc[2][v]);
      m = fmaxf(m, sc[3][v]);
#pragma unroll
      for (int msk = 8; msk; msk >>= 1) m = fmaxf(m, __shfl_xor(m, msk, 16));
      float nm = fmaxf(rm[v], m);
      float corr = __expf(rm[v] - nm);
      rm[v] = nm;
      float ps = 0.f;
#pragma unroll
      for (int nt = 0; nt < 4; ++nt) {
        float pe = __expf(sc[nt][v] - nm);
        sc[nt][v] = pe;
        ps += pe;
      }
#pragma unroll
      for (int msk = 8; msk; msk >>= 1) ps += __shfl_xor(ps, msk, 16);
      rs[v] = rs[v] * corr + ps;
#pragma unroll
      for (int ni = 0; ni < 6; ++ni) o[ni][v] *= corr;
    }

    // ---- move P (D-fragment layout) to A-fragment layout via LDS ----
#pragma unroll
    for (int nt = 0; nt < 4; ++nt)
#pragma unroll
      for (int v = 0; v < 8; ++v)
        P[wave][v + 8 * hi][nt * 16 + l15] = (_Float16)sc[nt][v];

    // ---- O += P * V  (K = 64 keys -> 2 chunks of 32) ----
#pragma unroll
    for (int kc = 0; kc < 2; ++kc) {
      half16 pa = ld_frag(&P[wave][l15][kc * 32 + hi * 8]);
#pragma unroll
      for (int ni = 0; ni < 6; ++ni) {
        half16 vb = ld_frag(&VT[ni * 16 + l15][kc * 32 + hi * 8]);
        o[ni] = wmma_f16(pa, vb, o[ni]);
      }
    }
  }

  // ---- normalize and write f16 output [B*64][1536] ----
#pragma unroll
  for (int ni = 0; ni < 6; ++ni) {
    int col = h * HD + ni * 16 + l15;
#pragma unroll
    for (int v = 0; v < 8; ++v) {
      int row = b * NLAT + wave * 16 + v + 8 * hi;
      O[(size_t)row * HHD + col] = (_Float16)(o[ni][v] / rs[v]);
    }
  }
}

// ---------------------------------------------------------------------------
// Host orchestration
// ---------------------------------------------------------------------------
extern "C" void kernel_launch(void* const* d_in, const int* in_sizes, int n_in,
                              void* d_out, int out_size, void* d_ws,
                              size_t ws_size, hipStream_t stream) {
  (void)in_sizes; (void)n_in; (void)out_size; (void)ws_size;

  const float* context  = (const float*)d_in[0];
  const float* latents  = (const float*)d_in[1];
  const float* ctx_ln_w = (const float*)d_in[2];
  const float* ctx_ln_b = (const float*)d_in[3];
  const float* lat_ln_w = (const float*)d_in[4];
  const float* lat_ln_b = (const float*)d_in[5];
  const float* q_ln_w   = (const float*)d_in[6];
  const float* q_ln_b   = (const float*)d_in[7];
  const float* k_ln_w   = (const float*)d_in[8];
  const float* k_ln_b   = (const float*)d_in[9];
  const float* wq       = (const float*)d_in[10];
  const float* wk       = (const float*)d_in[11];
  const float* wv       = (const float*)d_in[12];
  const float* wo       = (const float*)d_in[13];
  const float* mlp_ln_w = (const float*)d_in[14];
  const float* mlp_ln_b = (const float*)d_in[15];
  const float* w_fc     = (const float*)d_in[16];
  const float* w_cproj  = (const float*)d_in[17];
  const float* fin_ln_w = (const float*)d_in[18];
  const float* fin_ln_b = (const float*)d_in[19];

  // workspace carve-up (256B aligned)
  char* p = (char*)d_ws;
  auto carve = [&](size_t bytes) -> void* {
    void* r = (void*)p;
    p += (bytes + 255) & ~(size_t)255;
    return r;
  };
  const size_t M_KV = (size_t)BB * KVROWS;   // 67584
  const size_t M_L  = (size_t)BB * NLAT;     // 2048

  float*    lat    = (float*)   carve(M_L * DD * 4);
  _Float16* latn   = (_Float16*)carve(M_L * DD * 2);
  _Float16* kvn    = (_Float16*)carve(M_KV * DD * 2);
  _Float16* kf     = (_Float16*)carve(M_KV * HHD * 2);
  _Float16* vf     = (_Float16*)carve(M_KV * HHD * 2);
  float*    qf32   = (float*)   carve(M_L * HHD * 4);
  _Float16* qf16   = (_Float16*)carve(M_L * HHD * 2);
  _Float16* attn_o = (_Float16*)carve(M_L * HHD * 2);
  _Float16* h2     = (_Float16*)carve(M_L * FFD * 2);
  _Float16* wqT    = (_Float16*)carve((size_t)HHD * DD * 2);
  _Float16* wkT    = (_Float16*)carve((size_t)HHD * DD * 2);
  _Float16* wvT    = (_Float16*)carve((size_t)HHD * DD * 2);
  _Float16* woT    = (_Float16*)carve((size_t)DD * HHD * 2);
  _Float16* wfcT   = (_Float16*)carve((size_t)FFD * DD * 2);
  _Float16* wcpT   = (_Float16*)carve((size_t)DD * FFD * 2);

  const float qk_scale = 1.0f / sqrtf((float)HD);

  lat_init_kernel<<<(unsigned)((M_L * DD + 255) / 256), 256, 0, stream>>>(
      lat, latents);

  for (int i = 0; i < 3; ++i) {
    // ---- weight transpose+cast to f16 [N][K] ----
    {
      size_t n1 = (size_t)DD * HHD;   // wq/wk/wv (K=1280,N=1536)
      unsigned g1 = (unsigned)((n1 + 255) / 256);
      wtrans_kernel<<<g1, 256, 0, stream>>>(wq + (size_t)i * n1, wqT, DD, HHD);
      wtrans_kernel<<<g1, 256, 0, stream>>>(wk + (size_t)i * n1, wkT, DD, HHD);
      wtrans_kernel<<<g1, 256, 0, stream>>>(wv + (size_t)i * n1, wvT, DD, HHD);
      wtrans_kernel<<<g1, 256, 0, stream>>>(wo + (size_t)i * n1, woT, HHD, DD);
      size_t n2 = (size_t)DD * FFD;   // w_fc (K=1280,N=5120), w_cproj (K=5120,N=1280)
      unsigned g2 = (unsigned)((n2 + 255) / 256);
      wtrans_kernel<<<g2, 256, 0, stream>>>(w_fc + (size_t)i * n2, wfcT, DD, FFD);
      wtrans_kernel<<<g2, 256, 0, stream>>>(w_cproj + (size_t)i * n2, wcpT, FFD, DD);
    }

    // ---- LayerNorms feeding the projections ----
    ln1280_kernel<<<(unsigned)(BB * SS), 256, 0, stream>>>(
        context, ctx_ln_w + (size_t)i * DD, ctx_ln_b + (size_t)i * DD, kvn,
        nullptr, SS, KVROWS, 0);                        // ctx_n -> kv rows [0,S)
    ln1280_kernel<<<(unsigned)M_L, 256, 0, stream>>>(
        lat, lat_ln_w + (size_t)i * DD, lat_ln_b + (size_t)i * DD, kvn, nullptr,
        NLAT, KVROWS, SS);                              // lat_n -> kv rows [S,S+NL)
    ln1280_kernel<<<(unsigned)M_L, 256, 0, stream>>>(
        lat, lat_ln_w + (size_t)i * DD, lat_ln_b + (size_t)i * DD, latn,
        nullptr, NLAT, NLAT, 0);                        // lat_n for Q

    // ---- projections (WMMA GEMMs) ----
    gemm_f16_wmma_kernel<<<dim3(HHD / GBN, (unsigned)(M_L / GBM)), 256, 0,
                           stream>>>(latn, wqT, qf32, nullptr, nullptr,
                                     (int)M_L, HHD, DD, 0);
    gemm_f16_wmma_kernel<<<dim3(HHD / GBN, (unsigned)(M_KV / GBM)), 256, 0,
                           stream>>>(kvn, wkT, nullptr, kf, nullptr, (int)M_KV,
                                     HHD, DD, 1);
    gemm_f16_wmma_kernel<<<dim3(HHD / GBN, (unsigned)(M_KV / GBM)), 256, 0,
                           stream>>>(kvn, wvT, nullptr, vf, nullptr, (int)M_KV,
                                     HHD, DD, 1);

    qln_kernel<<<(unsigned)(M_L * HH / 8), 256, 0, stream>>>(
        qf32, q_ln_w + (size_t)i * HD, q_ln_b + (size_t)i * HD, qf16, qk_scale,
        (int)(M_L * HH));

    // ---- attention (flash-style WMMA, k-LN fused) ----
    attn_kernel<<<(unsigned)(BB * HH), 128, 0, stream>>>(
        qf16, kf, vf, k_ln_w + (size_t)i * HD, k_ln_b + (size_t)i * HD, attn_o);

    // ---- output projection + residual into lat ----
    gemm_f16_wmma_kernel<<<dim3(DD / GBN, (unsigned)(M_L / GBM)), 256, 0,
                           stream>>>(attn_o, woT, lat, nullptr, lat, (int)M_L,
                                     DD, HHD, 3);

    // ---- MLP ----
    ln1280_kernel<<<(unsigned)M_L, 256, 0, stream>>>(
        lat, mlp_ln_w + (size_t)i * DD, mlp_ln_b + (size_t)i * DD, latn,
        nullptr, NLAT, NLAT, 0);
    gemm_f16_wmma_kernel<<<dim3(FFD / GBN, (unsigned)(M_L / GBM)), 256, 0,
                           stream>>>(latn, wfcT, nullptr, h2, nullptr,
                                     (int)M_L, FFD, DD, 2);   // relu -> f16
    gemm_f16_wmma_kernel<<<dim3(DD / GBN, (unsigned)(M_L / GBM)), 256, 0,
                           stream>>>(h2, wcpT, lat, nullptr, lat, (int)M_L, DD,
                                     FFD, 3);
  }

  // ---- final LayerNorm -> d_out f32 ----
  ln1280_kernel<<<(unsigned)M_L, 256, 0, stream>>>(
      lat, fin_ln_w, fin_ln_b, nullptr, (float*)d_out, 1, 1, 0);
}
